// MultiHeadSelfAttention_47622597378334
// MI455X (gfx1250) — matrix operand order repaired and checked by reading, not compiled, not run on hardware
//
#include <hip/hip_runtime.h>
#include <hip/hip_bf16.h>

// ---------- vector types ----------
typedef __attribute__((ext_vector_type(16))) _Float16 v16h;
typedef __attribute__((ext_vector_type(8)))  _Float16 v8h;
typedef __attribute__((ext_vector_type(4)))  _Float16 v4h;
typedef __attribute__((ext_vector_type(8)))  float    v8f;
typedef __attribute__((ext_vector_type(4)))  float    v4f;

__device__ __forceinline__ v8f vzero8() {
  v8f z;
#pragma unroll
  for (int i = 0; i < 8; i++) z[i] = 0.0f;
  return z;
}

__device__ __forceinline__ v16h cat8(v8h lo, v8h hi) {
  v16h r;
#pragma unroll
  for (int i = 0; i < 8; i++) { r[i] = lo[i]; r[8 + i] = hi[i]; }
  return r;
}

// A-matrix fragment (16x32 f16), row-major source. p = start of this lane's row
// (m = lane&15) within the 32-wide K window, g = lane>>4.
// ISA layout: halves 0..7 = K 8g..8g+7 ; halves 8..15 = K 16+8g..16+8g+7.
__device__ __forceinline__ v16h frag_a_rowmajor(const _Float16* p, int g) {
  v8h lo = *(const v8h*)(p + 8 * g);
  v8h hi = *(const v8h*)(p + 16 + 8 * g);
  return cat8(lo, hi);
}

// B-matrix fragment (32x16 f16), column-contiguous source. p = start of this
// lane's column (n = lane&15) within the 32-wide K window, g = lane>>4.
// ISA layout: halves 0..7 = K 8g..8g+7 ; halves 8..15 = K 16+8g..16+8g+7.
__device__ __forceinline__ v16h frag_b_colmajor(const _Float16* p, int g) {
  v8h lo = *(const v8h*)(p + 8 * g);
  v8h hi = *(const v8h*)(p + 16 + 8 * g);
  return cat8(lo, hi);
}

// Async global -> LDS copy of 16 bytes (GVS mode: sgpr base + vgpr offset).
// Tracked by ASYNCcnt; lds_off is the 32-bit LDS byte offset (generic shared
// pointers truncate to the LDS offset per the aperture rules).
__device__ __forceinline__ void async_b128(uint32_t lds_off, const void* sbase,
                                           uint32_t goff_bytes) {
  asm volatile("global_load_async_to_lds_b128 %0, %1, %2"
               :: "v"(lds_off), "v"(goff_bytes), "s"(sbase) : "memory");
}

// ============================================================================
// Pre-pass: f32 -> f16 copy (x), vectorized 4-wide.
// ============================================================================
__global__ __launch_bounds__(256) void cvt_f16(const float* __restrict__ in,
                                               _Float16* __restrict__ out,
                                               int n4) {
  int i = blockIdx.x * 256 + threadIdx.x;
  if (i < n4) {
    v4f v = *(const v4f*)(in + (size_t)i * 4);
    v4h h;
#pragma unroll
    for (int e = 0; e < 4; e++) h[e] = (_Float16)v[e];
    *(v4h*)(out + (size_t)i * 4) = h;
  }
}

// ============================================================================
// Pre-pass: W[K,N] f32 -> Wt[N,K] f16 (LDS-tiled transpose, 32x32 tiles).
// ============================================================================
__global__ __launch_bounds__(256) void transpose_cvt(
    const float* __restrict__ W, _Float16* __restrict__ Wt, int K, int N) {
  __shared__ float tile[32][33];
  const int tx = threadIdx.x & 31;
  const int ty = threadIdx.x >> 5;  // 0..7
  const int n0 = blockIdx.x * 32;
  const int k0 = blockIdx.y * 32;
#pragma unroll
  for (int i = 0; i < 4; i++)
    tile[ty + 8 * i][tx] = W[(size_t)(k0 + ty + 8 * i) * N + n0 + tx];
  __syncthreads();
#pragma unroll
  for (int i = 0; i < 4; i++)
    Wt[(size_t)(n0 + ty + 8 * i) * K + k0 + tx] = (_Float16)tile[tx][ty + 8 * i];
}

// ============================================================================
// Double-buffered async GEMM: C[M,N] = A[M,K] * Bt[N,K]^T, all-f16 sources,
// f32 WMMA accumulate. Block = 256 threads (8 waves), C tile 128x128, BK = 32.
// Tiles stream via global_load_async_to_lds_b128 (4 async insts per wave per
// tile; s_wait_asynccnt 4 keeps exactly one tile prefetched).
// QKV_EP: write f16 with 1/sqrt(64) scaling on Q columns (col%192 < 64);
//         else write f32.
// ============================================================================
template <bool QKV_EP>
__global__ __launch_bounds__(256) void gemm_wmma_async(
    const _Float16* __restrict__ A, const _Float16* __restrict__ Bt,
    void* __restrict__ Cp, int M, int N, int K) {
  __shared__ _Float16 As[2][128 * 40];  // row-major, stride 40 halves
  __shared__ _Float16 Bs[2][128 * 40];  // Bt rows = B columns, stride 40

  const int t = threadIdx.x;
  const int lane = t & 31;
  const int w = t >> 5;
  const int wm = (w >> 2) * 64;
  const int wn = (w & 3) * 32;
  const int m0 = blockIdx.y * 128;
  const int n0 = blockIdx.x * 128;
  const int nlane = lane & 15;
  const int g = lane >> 4;

  // Issue one K-tile (128x32 of A and of Bt) as async b128 copies.
  auto issue = [&](int buf, int k0) {
#pragma unroll
    for (int i = 0; i < 2; i++) {
      int c = t + i * 256;      // 512 chunks of 8 halves each
      int row = c >> 2;
      int kc = (c & 3) * 8;
      uint32_t lA = (uint32_t)(uintptr_t)&As[buf][row * 40 + kc];
      uint32_t gA = (uint32_t)(((size_t)(m0 + row) * K + k0 + kc) * 2);
      async_b128(lA, A, gA);
      uint32_t lB = (uint32_t)(uintptr_t)&Bs[buf][row * 40 + kc];
      uint32_t gB = (uint32_t)(((size_t)(n0 + row) * K + k0 + kc) * 2);
      async_b128(lB, Bt, gB);
    }
  };

  v8f acc[4][2];
#pragma unroll
  for (int i = 0; i < 4; i++)
#pragma unroll
    for (int j = 0; j < 2; j++) acc[i][j] = vzero8();

  const int NT = K / 32;
  issue(0, 0);
  for (int kk = 0; kk < NT; kk++) {
    const int buf = kk & 1;
    if (kk + 1 < NT) {
      issue(buf ^ 1, (kk + 1) * 32);
      asm volatile("s_wait_asynccnt 4" ::: "memory");
    } else {
      asm volatile("s_wait_asynccnt 0" ::: "memory");
    }
    __syncthreads();  // this tile's data visible to all waves

    v16h af[4], bf[2];
#pragma unroll
    for (int mt = 0; mt < 4; mt++)
      af[mt] = frag_a_rowmajor(As[buf] + (wm + mt * 16 + nlane) * 40, g);
#pragma unroll
    for (int nt = 0; nt < 2; nt++)
      bf[nt] = frag_b_colmajor(Bs[buf] + (wn + nt * 16 + nlane) * 40, g);
#pragma unroll
    for (int mt = 0; mt < 4; mt++)
#pragma unroll
      for (int nt = 0; nt < 2; nt++)
        acc[mt][nt] = __builtin_amdgcn_wmma_f32_16x16x32_f16(
            false, af[mt], false, bf[nt], (short)0, acc[mt][nt], false, false);
    __syncthreads();  // done reading this buffer before it is refilled
  }

  // ---- epilogue (D layout: lane holds col n = lane&15, rows 8g+r) ----
#pragma unroll
  for (int mt = 0; mt < 4; mt++) {
#pragma unroll
    for (int nt = 0; nt < 2; nt++) {
      int col = n0 + wn + nt * 16 + nlane;
#pragma unroll
      for (int r = 0; r < 8; r++) {
        int row = m0 + wm + mt * 16 + 8 * g + r;
        float val = acc[mt][nt][r];
        if (QKV_EP) {
          _Float16* C16 = (_Float16*)Cp;
          float s = ((col % 192) < 64) ? 0.125f : 1.0f;  // q / sqrt(64)
          C16[(size_t)row * N + col] = (_Float16)(val * s);
        } else {
          float* C32 = (float*)Cp;
          C32[(size_t)row * N + col] = val;
        }
      }
    }
  }
}

// ============================================================================
// Flash attention: one block per (b, h, 64-row query block). 128 threads
// (4 waves); each wave owns 16 query rows. Online softmax, f32 accumulators.
// qkv layout per row: [q(64) k(64) v(64)] per head, row stride 3072 halves.
// K chunk staged with async copies; V transposed synchronously.
// ============================================================================
__global__ __launch_bounds__(128) void attn_wmma(
    const _Float16* __restrict__ qkv, const float* __restrict__ rpe,
    _Float16* __restrict__ att) {
  constexpr int L = 1024, H = 16;
  __shared__ _Float16 Ks[64 * 72];       // K chunk row-major [jrow][d]
  __shared__ _Float16 Vt[64 * 72];       // V chunk transposed [d][jrow]
  __shared__ float bias_s[128];          // 127 diagonals of rpe bias
  __shared__ _Float16 Ps[4][16 * 72];    // per-wave P staging (D->A relayout)

  const int t = threadIdx.x;
  const int lane = t & 31;
  const int w = t >> 5;
  const int nlane = lane & 15;
  const int g = lane >> 4;

  const int qblk = blockIdx.x & 15;
  const int bh = blockIdx.x >> 4;
  const int h = bh & 15;
  const int b = bh >> 4;
  const int q0 = qblk * 64;

  const _Float16* base = qkv + (size_t)b * L * 3072 + h * 192;

  // Q fragments: direct strided global loads (row = q0 + w*16 + nlane)
  v16h qf[2];
  {
    const _Float16* qrow = base + (size_t)(q0 + w * 16 + nlane) * 3072;
#pragma unroll
    for (int ks = 0; ks < 2; ks++) qf[ks] = frag_a_rowmajor(qrow + ks * 32, g);
  }

  float rowm[8], rowl[8];
#pragma unroll
  for (int r = 0; r < 8; r++) { rowm[r] = -1e30f; rowl[r] = 0.0f; }
  v8f acco[4];
#pragma unroll
  for (int i = 0; i < 4; i++) acco[i] = vzero8();

  for (int jc = 0; jc < 16; jc++) {
    const int j0 = jc * 64;
    // ---- async-stage K chunk (row-major, b128 chunks) ----
#pragma unroll
    for (int i = 0; i < 4; i++) {
      int c = t + i * 128;              // 512 chunks of 8 halves
      int row = c >> 3;
      int dc = (c & 7) * 8;
      uint32_t l = (uint32_t)(uintptr_t)&Ks[row * 72 + dc];
      uint32_t go = (uint32_t)(((size_t)(j0 + row) * 3072 + 64 + dc) * 2);
      async_b128(l, base, go);
    }
    // ---- stage V chunk transposed (overlaps with async K) ----
#pragma unroll
    for (int i = 0; i < 4; i++) {
      int c = t + i * 128;
      int row = c >> 3;
      int dc = (c & 7) * 8;
      v8h v = *(const v8h*)(base + (size_t)(j0 + row) * 3072 + 128 + dc);
#pragma unroll
      for (int e = 0; e < 8; e++) Vt[(dc + e) * 72 + row] = v[e];
    }
    // ---- stage the 127 bias diagonals for this (q-block, j-chunk) ----
    if (t < 127) {
      int off = (q0 - j0) + (t - 63) + (L - 1);   // in [0, 2046]
      bias_s[t] = rpe[off * H + h];
    }
    asm volatile("s_wait_asynccnt 0" ::: "memory");
    __syncthreads();

    // ---- S = Q * K^T  (wave's 16 rows x 64 key cols) ----
    v8f accs[4];
#pragma unroll
    for (int nt = 0; nt < 4; nt++) accs[nt] = vzero8();
#pragma unroll
    for (int nt = 0; nt < 4; nt++) {
#pragma unroll
      for (int ks = 0; ks < 2; ks++) {
        v16h bf = frag_b_colmajor(Ks + (nt * 16 + nlane) * 72 + ks * 32, g);
        accs[nt] = __builtin_amdgcn_wmma_f32_16x16x32_f16(
            false, qf[ks], false, bf, (short)0, accs[nt], false, false);
      }
    }

    // ---- bias + online softmax (rows partitioned: group g owns rows 8g+r) --
    float p[4][8];
#pragma unroll
    for (int nt = 0; nt < 4; nt++)
#pragma unroll
      for (int r = 0; r < 8; r++) {
        int x = w * 16 + 8 * g + r;   // row in query block
        int y = nt * 16 + nlane;      // col (key) in chunk
        p[nt][r] = accs[nt][r] + bias_s[x - y + 63];
      }
#pragma unroll
    for (int r = 0; r < 8; r++) {
      float mx = p[0][r];
#pragma unroll
      for (int nt = 1; nt < 4; nt++) mx = fmaxf(mx, p[nt][r]);
#pragma unroll
      for (int off = 1; off < 16; off <<= 1)
        mx = fmaxf(mx, __shfl_xor(mx, off, 16));
      float nm = fmaxf(rowm[r], mx);
      float corr = __expf(rowm[r] - nm);
      float sum = 0.0f;
#pragma unroll
      for (int nt = 0; nt < 4; nt++) {
        p[nt][r] = __expf(p[nt][r] - nm);
        sum += p[nt][r];
      }
#pragma unroll
      for (int off = 1; off < 16; off <<= 1) sum += __shfl_xor(sum, off, 16);
      rowl[r] = rowl[r] * corr + sum;
      rowm[r] = nm;
#pragma unroll
      for (int nt = 0; nt < 4; nt++) acco[nt][r] *= corr;
    }

    // ---- relayout P (D-layout -> A-layout) through per-wave LDS ----
#pragma unroll
    for (int nt = 0; nt < 4; nt++)
#pragma unroll
      for (int r = 0; r < 8; r++)
        Ps[w][(8 * g + r) * 72 + nt * 16 + nlane] = (_Float16)p[nt][r];
    // same-wave LDS RAW: DS ops are in-order per wave; fence the compiler.
    asm volatile("s_wait_dscnt 0" ::: "memory");

    // ---- O += P * V ----
#pragma unroll
    for (int ks = 0; ks < 2; ks++) {
      v16h pa = frag_a_rowmajor(Ps[w] + nlane * 72 + ks * 32, g);
#pragma unroll
      for (int nt = 0; nt < 4; nt++) {
        v16h bf = frag_b_colmajor(Vt + (nt * 16 + nlane) * 72 + ks * 32, g);
        acco[nt] = __builtin_amdgcn_wmma_f32_16x16x32_f16(
            false, pa, false, bf, (short)0, acco[nt], false, false);
      }
    }
    __syncthreads();  // before next chunk overwrites Ks/Vt/bias
  }

  // ---- normalize + write f16 attention output [B, L, F] ----
#pragma unroll
  for (int nt = 0; nt < 4; nt++) {
#pragma unroll
    for (int r = 0; r < 8; r++) {
      int i = q0 + w * 16 + 8 * g + r;
      int col = h * 64 + nt * 16 + nlane;
      att[((size_t)b * L + i) * 1024 + col] = (_Float16)(acco[nt][r] / rowl[r]);
    }
  }
}

// ============================================================================
extern "C" void kernel_launch(void* const* d_in, const int* in_sizes, int n_in,
                              void* d_out, int out_size, void* d_ws,
                              size_t ws_size, hipStream_t stream) {
  const float* x     = (const float*)d_in[0];   // [8,1024,1024]
  const float* w_qkv = (const float*)d_in[1];   // [1024,3072]
  const float* rpe   = (const float*)d_in[2];   // [2047,16]
  const float* w_out = (const float*)d_in[3];   // [1024,1024]
  float* out = (float*)d_out;                   // [8,1024,1024] f32

  // workspace layout (bytes)
  char* ws = (char*)d_ws;
  _Float16* qkv16 = (_Float16*)(ws);                         // 8192x3072 (48MB)
  _Float16* att16 = (_Float16*)(ws + 50331648);              // 8192x1024 (16MB)
  _Float16* x16   = (_Float16*)(ws + 67108864);              // 8192x1024 (16MB)
  _Float16* wqkvT = (_Float16*)(ws + 83886080);              // 3072x1024 (6MB)
  _Float16* woutT = (_Float16*)(ws + 90177536);              // 1024x1024 (2MB)

  // 0) precision / layout pre-passes
  cvt_f16<<<8192, 256, 0, stream>>>(x, x16, 8 * 1024 * 1024 / 4);
  transpose_cvt<<<dim3(3072 / 32, 1024 / 32), 256, 0, stream>>>(w_qkv, wqkvT,
                                                                1024, 3072);
  transpose_cvt<<<dim3(1024 / 32, 1024 / 32), 256, 0, stream>>>(w_out, woutT,
                                                                1024, 1024);

  // 1) QKV projection: [8192,1024] x [1024,3072] -> f16, q-scaled
  dim3 g1(3072 / 128, 8192 / 128);
  gemm_wmma_async<true><<<g1, 256, 0, stream>>>(x16, wqkvT, (void*)qkv16, 8192,
                                                3072, 1024);

  // 2) flash attention per (b,h,qblock)
  attn_wmma<<<8 * 16 * 16, 128, 0, stream>>>(qkv16, rpe, att16);

  // 3) output projection: [8192,1024](f16) x [1024,1024] -> f32
  dim3 g2(1024 / 128, 8192 / 128);
  gemm_wmma_async<false><<<g2, 256, 0, stream>>>(att16, woutT, (void*)out, 8192,
                                                 1024, 1024);
}
